// Model_78228534330127
// MI455X (gfx1250) — matrix operand order repaired and checked by reading, not compiled, not run on hardware
//
#include <hip/hip_runtime.h>

#define T_STEPS 8192
#define DIM 2048
#define NBLK_SCAN 64
#define ROWS_PER_BLK (DIM / NBLK_SCAN)   // 32 rows of W2 per workgroup, 256 KB LDS

typedef __attribute__((ext_vector_type(16))) _Float16 v16h;
typedef __attribute__((ext_vector_type(8)))  float    v8f;
typedef __attribute__((ext_vector_type(4)))  float    v4f;

// ---------------------------------------------------------------------------
// Phase 1: pre = x @ W1^T + b1   (8192x2048 @ 2048x2048) via WMMA f16->f32.
// One wave computes a 16x64 output strip: A tile loaded once per K-step,
// reused across 4 N-subtiles. Layouts follow CDNA5 ISA:
//   A 16x32 f16: lane m<16 holds K {0..7,16..23}; lane m+16 holds K {8..15,24..31}
//   B 32x16 f16: lane n<16 holds K {0..7,16..23}; lane n+16 holds K {8..15,24..31} (col n)
//   C/D f32:     lane<16 -> N=lane, M=r; lane>=16 -> N=lane-16, M=8+r
// ---------------------------------------------------------------------------
__global__ __launch_bounds__(256) void fc1_gemm(const float* __restrict__ x,
                                                const float* __restrict__ W1,
                                                const float* __restrict__ b1,
                                                float* __restrict__ pre) {
  const int lane = threadIdx.x & 31;
  const int wid  = threadIdx.x >> 5;
  const int unit = blockIdx.x * 8 + wid;     // 16384 wave-units
  const int mtile = unit >> 5;               // 512 M-tiles of 16 rows
  const int ngrp  = unit & 31;               // 32 N-groups of 64 cols
  const int t0 = mtile * 16;
  const int j0 = ngrp * 64;
  const int lm = lane & 15;
  const int hi = lane >> 4;

  const float* xrow = x + (size_t)(t0 + lm) * DIM;
  v8f acc[4] = {};

  for (int k0 = 0; k0 < DIM; k0 += 32) {
    // ---- A tile (f32 -> f16) ----
    const int ka = k0 + hi * 8;
    v4f a0 = *(const v4f*)(xrow + ka);
    v4f a1 = *(const v4f*)(xrow + ka + 4);
    v4f a2 = *(const v4f*)(xrow + ka + 16);
    v4f a3 = *(const v4f*)(xrow + ka + 20);
    v16h a;
#pragma unroll
    for (int i = 0; i < 4; ++i) {
      a[i]      = (_Float16)a0[i];
      a[4 + i]  = (_Float16)a1[i];
      a[8 + i]  = (_Float16)a2[i];
      a[12 + i] = (_Float16)a3[i];
    }
    // ---- 4 B tiles + WMMA ----
#pragma unroll
    for (int s = 0; s < 4; ++s) {
      const float* wrow = W1 + (size_t)(j0 + s * 16 + lm) * DIM + k0 + hi * 8;
      v4f w0 = *(const v4f*)(wrow);
      v4f w1 = *(const v4f*)(wrow + 4);
      v4f w2 = *(const v4f*)(wrow + 16);
      v4f w3 = *(const v4f*)(wrow + 20);
      v16h b;
#pragma unroll
      for (int i = 0; i < 4; ++i) {
        b[i]      = (_Float16)w0[i];
        b[4 + i]  = (_Float16)w1[i];
        b[8 + i]  = (_Float16)w2[i];
        b[12 + i] = (_Float16)w3[i];
      }
      acc[s] = __builtin_amdgcn_wmma_f32_16x16x32_f16(
          /*neg_a=*/false, a, /*neg_b=*/false, b,
          /*c_mod=*/(short)0, acc[s], /*reuse_a=*/false, /*reuse_b=*/false);
    }
  }

#pragma unroll
  for (int s = 0; s < 4; ++s) {
    const int j = j0 + s * 16 + lm;
    const float bb = b1[j];
#pragma unroll
    for (int r = 0; r < 8; ++r) {
      const int t = t0 + hi * 8 + r;
      pre[(size_t)t * DIM + j] = acc[s][r] + bb;
    }
  }
}

// ---------------------------------------------------------------------------
// Init: seed h ping-pong buffer with h0, reset grid-barrier counters.
// Runs every kernel_launch so graph replays start from a clean state.
// ---------------------------------------------------------------------------
__global__ void scan_init(const float* __restrict__ h0, float* __restrict__ hA,
                          unsigned* __restrict__ sync) {
  const int tid = threadIdx.x;
  for (int k = tid; k < DIM; k += 256) hA[k] = h0[k];
  if (tid < 8) sync[tid] = 0u;
}

// ---------------------------------------------------------------------------
// Phase 2+3: persistent scan. 64 WGs, each holds 32 rows of W2 (fp32) in LDS
// (256 KB — only possible on CDNA5's 320 KB/WGP LDS). Per step: load h (8 KB)
// to LDS, each wave dots 4 LDS-resident rows with float4 ds loads, wave-shfl
// reduce, relu+store, then a generation grid barrier over L2 atomics.
// Block 0 finishes with out = h @ Wf^T + bf.
// ---------------------------------------------------------------------------
__global__ __launch_bounds__(256) void rnn_scan(const float* __restrict__ pre,
                                                const float* __restrict__ W2,
                                                const float* __restrict__ b2,
                                                const float* __restrict__ Wf,
                                                const float* __restrict__ bf,
                                                float* __restrict__ hA,
                                                float* __restrict__ hB,
                                                unsigned* __restrict__ sync, // [0]=cnt [1]=gen
                                                float* __restrict__ out) {
  extern __shared__ float smem[];
  float* w2s = smem;                        // ROWS_PER_BLK * DIM
  float* hl  = smem + ROWS_PER_BLK * DIM;   // DIM

  const int tid  = threadIdx.x;
  const int lane = tid & 31;
  const int wid  = tid >> 5;
  const int j0g  = blockIdx.x * ROWS_PER_BLK;

  // Stage this WG's 32 contiguous W2 rows into LDS (coalesced 256 KB copy).
  for (int idx = tid; idx < ROWS_PER_BLK * DIM; idx += 256)
    w2s[idx] = W2[(size_t)j0g * DIM + idx];

  float b2r[4];
#pragma unroll
  for (int rr = 0; rr < 4; ++rr) b2r[rr] = b2[j0g + wid * 4 + rr];

  __syncthreads();

  for (int t = 0; t < T_STEPS; ++t) {
    const float* hsrc = (t & 1) ? hB : hA;
    float*       hdst = (t & 1) ? hA : hB;

    for (int k = tid; k < DIM; k += 256) hl[k] = hsrc[k];
    if (tid == 0 && t + 1 < T_STEPS)
      __builtin_prefetch(&pre[(size_t)(t + 1) * DIM + j0g], 0, 0);
    __syncthreads();

    const v4f* h4 = (const v4f*)hl;
#pragma unroll
    for (int rr = 0; rr < 4; ++rr) {
      const int jl = wid * 4 + rr;
      const v4f* w4 = (const v4f*)(w2s + jl * DIM);
      float s = 0.f;
#pragma unroll 4
      for (int k = lane; k < DIM / 4; k += 32) {
        v4f wv = w4[k];
        v4f hv = h4[k];
        s += wv.x * hv.x + wv.y * hv.y + wv.z * hv.z + wv.w * hv.w;
      }
#pragma unroll
      for (int off = 16; off > 0; off >>= 1) s += __shfl_down(s, off, 32);
      if (lane == 0) {
        float v = pre[(size_t)t * DIM + j0g + jl] + s + b2r[rr];
        hdst[j0g + jl] = v > 0.f ? v : 0.f;
      }
    }

    // ---- grid barrier (generation counter over device-scope atomics) ----
    __syncthreads();
    if (tid == 0) {
      __threadfence();
      unsigned g = __hip_atomic_load(&sync[1], __ATOMIC_RELAXED, __HIP_MEMORY_SCOPE_AGENT);
      unsigned a = __hip_atomic_fetch_add(&sync[0], 1u, __ATOMIC_ACQ_REL, __HIP_MEMORY_SCOPE_AGENT);
      if (a == NBLK_SCAN - 1) {
        __hip_atomic_store(&sync[0], 0u, __ATOMIC_RELAXED, __HIP_MEMORY_SCOPE_AGENT);
        __hip_atomic_fetch_add(&sync[1], 1u, __ATOMIC_RELEASE, __HIP_MEMORY_SCOPE_AGENT);
      } else {
        while (__hip_atomic_load(&sync[1], __ATOMIC_ACQUIRE, __HIP_MEMORY_SCOPE_AGENT) == g)
          __builtin_amdgcn_s_sleep(1);
      }
    }
    __syncthreads();
  }

  // ---- Phase 3: out = h @ Wf^T + bf (T even -> final h in hA) ----
  if (blockIdx.x == 0) {
    for (int j = 0; j < 4; ++j) {
      float s = 0.f;
      for (int k = tid; k < DIM; k += 256) s += hA[k] * Wf[j * DIM + k];
#pragma unroll
      for (int off = 16; off > 0; off >>= 1) s += __shfl_down(s, off, 32);
      if (lane == 0) hl[wid] = s;
      __syncthreads();
      if (tid == 0) {
        float tot = 0.f;
#pragma unroll
        for (int w = 0; w < 8; ++w) tot += hl[w];
        out[j] = tot + bf[j];
      }
      __syncthreads();
    }
  }
}

// ---------------------------------------------------------------------------
extern "C" void kernel_launch(void* const* d_in, const int* in_sizes, int n_in,
                              void* d_out, int out_size, void* d_ws, size_t ws_size,
                              hipStream_t stream) {
  const float* x  = (const float*)d_in[0];
  const float* h0 = (const float*)d_in[1];
  const float* W1 = (const float*)d_in[2];
  const float* b1 = (const float*)d_in[3];
  const float* W2 = (const float*)d_in[4];
  const float* b2 = (const float*)d_in[5];
  const float* Wf = (const float*)d_in[6];
  const float* bf = (const float*)d_in[7];
  float* out = (float*)d_out;

  char* ws = (char*)d_ws;
  float* pre = (float*)ws;                                       // T*D fp32 = 64 MB
  float* hA  = (float*)(ws + (size_t)T_STEPS * DIM * sizeof(float));
  float* hB  = hA + DIM;
  unsigned* sync = (unsigned*)(hB + DIM);

  // Phase 1: 16384 wave-tiles of 16x64 -> 2048 blocks x 8 waves.
  fc1_gemm<<<2048, 256, 0, stream>>>(x, W1, b1, pre);

  // Reset recurrent state + barrier counters every call (graph-replay safe).
  scan_init<<<1, 256, 0, stream>>>(h0, hA, sync);

  // Phase 2: persistent scan, 270 KB dynamic LDS per WG (needs CDNA5 320 KB).
  const size_t smem = (size_t)(ROWS_PER_BLK * DIM + DIM) * sizeof(float);
  hipFuncSetAttribute((const void*)rnn_scan,
                      hipFuncAttributeMaxDynamicSharedMemorySize, (int)smem);
  rnn_scan<<<NBLK_SCAN, 256, smem, stream>>>(pre, W2, b2, Wf, bf, hA, hB, sync, out);
}
